// mamba_86535001080292
// MI455X (gfx1250) — compile-verified, matrix operand order checked by reading, not run on hardware
//
#include <hip/hip_runtime.h>
#include <cstdint>

typedef __attribute__((ext_vector_type(16))) _Float16 v16h;
typedef __attribute__((ext_vector_type(8)))  _Float16 h8;
typedef __attribute__((ext_vector_type(8)))  float    v8f;

__device__ __forceinline__ float sigmoidf_(float x) { return 1.f / (1.f + __expf(-x)); }

// ---------------------------------------------------------------------------
// weight cast + transpose: wt[N,K] (f16, N-major) = w[K,N] (f32, K-major)
// makes the WMMA B-fragment a pair of contiguous b128 loads.
// ---------------------------------------------------------------------------
__global__ void k_wcast_t(const float* __restrict__ w, _Float16* __restrict__ wt,
                          int K, int N, long total) {
    long i = (long)blockIdx.x * blockDim.x + threadIdx.x;
    if (i >= total) return;
    const int  n = (int)(i % N);
    const long k = i / N;
    wt[(long)n * K + k] = (_Float16)w[i];
}

// ---------------------------------------------------------------------------
// WMMA GEMM: C[M, NT*16] strip = epi(A[M,K](f32) @ Bt[N,K](f16)^T + bias + res)
// Compile-time NT (1..4) -> branch-free inner loop. Per K-step: issue all NT
// B-fragment b128 load clauses first (vector loads land directly in the WMMA
// source tuples -> no repack moves, no WMMA->VALU hazard NOPs), then convert
// A once (packed cvt), then run NT v_wmma back-to-back with staggered waits.
// Fragment layout per CDNA5 ISA 7.12.2 (wave32): lane l (half=l/16, mr=l%16),
// halves e=0..7 -> k0+8*half+e, e=8..15 -> k0+16+8*half+e.
// act: 0 = none, 1 = softplus
// ---------------------------------------------------------------------------
template <int NT>
__global__ void k_wmma_gemm(const float* __restrict__ A, int lda,
                            const _Float16* __restrict__ Bt,
                            float* __restrict__ C, int ldc,
                            const float* __restrict__ bias,
                            const float* __restrict__ residual, int ldr,
                            int ncol0, int K, int act) {
    const int lane = threadIdx.x;
    const int half = lane >> 4;
    const int mr   = lane & 15;
    const long rowA = (long)blockIdx.x * 16 + mr;
    const int  n0   = ncol0 + blockIdx.y * 64;

    v8f acc[NT] = {};
    for (int k0 = 0; k0 < K; k0 += 32) {
        // stage all NT B fragments: exact b128 vector loads, no repacking
        v16h bf[NT];
#pragma unroll
        for (int j = 0; j < NT; ++j) {
            const h8* pb = (const h8*)(Bt + (long)(n0 + j * 16 + mr) * K + k0 + half * 8);
            const h8 lo = pb[0];
            const h8 hi = pb[2];  // +16 halves
            bf[j] = __builtin_shufflevector(lo, hi, 0, 1, 2, 3, 4, 5, 6, 7,
                                            8, 9, 10, 11, 12, 13, 14, 15);
        }
        // A fragment: two 32B vector loads, f32 -> f16 packed converts
        const float* paf = A + rowA * lda + k0 + half * 8;
        if (k0 + 32 < K) __builtin_prefetch(paf + 32, 0, 1);  // global_prefetch_b8
        const v8f alo = *(const v8f*)paf;
        const v8f ahi = *(const v8f*)(paf + 16);
        v16h a;
#pragma unroll
        for (int e = 0; e < 8; ++e) {
            a[e]     = (_Float16)alo[e];
            a[e + 8] = (_Float16)ahi[e];
        }
        // NT WMMAs back-to-back; waits stagger as bf[j] arrive
#pragma unroll
        for (int j = 0; j < NT; ++j) {
            acc[j] = __builtin_amdgcn_wmma_f32_16x16x32_f16(
                false, a, false, bf[j], (short)0, acc[j], false, false);
        }
    }

#pragma unroll
    for (int j = 0; j < NT; ++j) {
#pragma unroll
        for (int v = 0; v < 8; ++v) {
            const long m  = (long)blockIdx.x * 16 + v + half * 8;
            const int  nn = n0 + j * 16 + mr;
            float val = acc[j][v];
            if (bias)     val += bias[nn];
            if (residual) val += residual[m * ldr + nn];
            if (act == 1) val = (val > 20.f) ? val : log1pf(__expf(val));  // softplus
            C[m * ldc + nn] = val;
        }
    }
}

// ---------------------------------------------------------------------------
// depthwise causal conv (k=2) + SiLU:
//   xc[b,t,c] = silu(cb[c] + cw[c,0]*x[b,t-1,c] + cw[c,1]*x[b,t,c])
// x is the first-d columns of xz (row stride 2d).
// ---------------------------------------------------------------------------
__global__ void k_conv_silu(const float* __restrict__ xz, float* __restrict__ xc,
                            const float* __restrict__ cw, const float* __restrict__ cb,
                            int d, int L, long total) {
    long i = (long)blockIdx.x * blockDim.x + threadIdx.x;
    if (i >= total) return;
    const int  c   = (int)(i % d);
    const long row = i / d;
    const int  t   = (int)(row % L);
    const long ld2 = 2 * (long)d;
    const float cur  = xz[row * ld2 + c];
    const float prev = (t > 0) ? xz[(row - 1) * ld2 + c] : 0.f;
    const float v = cb[c] + cw[c * 2 + 0] * prev + cw[c * 2 + 1] * cur;
    xc[i] = v * sigmoidf_(v);
}

// ---------------------------------------------------------------------------
// selective scan: one block per (b, channel), n=128 states across 128 lanes
// (4 wave32s). The per-step B/C state rows are double-buffered through LDS
// with async global->LDS loads (CDNA5 ASYNCcnt path): step t+1 is in flight
// while step t computes.  Fuses: y = (scan + u*dskip) * silu(z)
// ---------------------------------------------------------------------------
__global__ void k_scan(const float* __restrict__ xc, const float* __restrict__ delta,
                       const float* __restrict__ xdbl, int ldx, int r,
                       const float* __restrict__ alog, const float* __restrict__ dskip,
                       const float* __restrict__ xz, int twoD,
                       float* __restrict__ ys, int d, int L, int n) {
    const int bd  = blockIdx.x;
    const int b   = bd / d;
    const int dch = bd % d;
    const int ni  = threadIdx.x;

    const float Av  = -__expf(alog[(long)dch * n + ni]);
    const float Dsk = dskip[dch];
    float h = 0.f;

    __shared__ float red[4];
    __shared__ float stage[2][2][128];  // [phase][B/C][n]

    auto issue = [&](int t, int p) {
        const long row = (long)b * L + t;
        const float* gB = xdbl + row * ldx + r + ni;
        const float* gC = gB + n;
        unsigned ldsB = (unsigned)(size_t)&stage[p][0][ni];
        unsigned ldsC = (unsigned)(size_t)&stage[p][1][ni];
        unsigned long long aB = (unsigned long long)(size_t)gB;
        unsigned long long aC = (unsigned long long)(size_t)gC;
        asm volatile("global_load_async_to_lds_b32 %0, %1, off"
                     :: "v"(ldsB), "v"(aB) : "memory");
        asm volatile("global_load_async_to_lds_b32 %0, %1, off"
                     :: "v"(ldsC), "v"(aC) : "memory");
    };

    issue(0, 0);
    int p = 0;
    for (int t = 0; t < L; ++t) {
        asm volatile("s_wait_asynccnt 0x0" ::: "memory");
        __syncthreads();  // all waves' async stages for phase p visible
        if (t + 1 < L) issue(t + 1, p ^ 1);

        const long row = (long)b * L + t;
        const float dt = delta[row * d + dch];
        const float u  = xc[row * d + dch];
        const float Bv = stage[p][0][ni];
        const float Cv = stage[p][1][ni];
        h = __expf(dt * Av) * h + (dt * u) * Bv;
        float pv = h * Cv;
#pragma unroll
        for (int o = 16; o > 0; o >>= 1) pv += __shfl_down(pv, o, 32);
        if ((ni & 31) == 0) red[ni >> 5] = pv;
        __syncthreads();
        if (ni == 0) {
            const float s  = red[0] + red[1] + red[2] + red[3] + u * Dsk;
            const float zv = xz[row * twoD + d + dch];
            ys[row * d + dch] = s * (zv * sigmoidf_(zv));
        }
        __syncthreads();
        p ^= 1;
    }
}

// ---------------------------------------------------------------------------
// final gating head: out[b] = houts[b,:] @ wg + bg   (B=32, K=128)
// ---------------------------------------------------------------------------
__global__ void k_final(const float* __restrict__ houts, const float* __restrict__ wg,
                        const float* __restrict__ bg, float* __restrict__ out,
                        int Bn, int Kn) {
    const int b = threadIdx.x;
    if (b < Bn) {
        float s = bg[0];
        for (int k = 0; k < Kn; ++k) s += houts[(long)b * Kn + k] * wg[k];
        out[b] = s;
    }
}

// ---------------------------------------------------------------------------
extern "C" void kernel_launch(void* const* d_in, const int* in_sizes, int n_in,
                              void* d_out, int out_size, void* d_ws, size_t ws_size,
                              hipStream_t stream) {
    (void)in_sizes; (void)n_in; (void)out_size; (void)ws_size;
    const int B = 32, L = 256, IN = 64, OUT = 64, NST = 128;
    const long M = (long)B * L;  // 8192 rows

    const float* x = (const float*)d_in[0];

    // bump-allocate workspace (sized for the d=1024 head, reused by both heads)
    char* ws = (char*)d_ws;
    size_t off = 0;
    auto alloc = [&](size_t bytes) -> void* {
        void* p = ws + off;
        off += (bytes + 255) & ~(size_t)255;
        return p;
    };
    float*    tmp   = (float*)alloc(M * 1024 * sizeof(float));
    float*    xz    = (float*)alloc(M * 2048 * sizeof(float));
    float*    xc    = (float*)alloc(M * 1024 * sizeof(float));
    float*    xdbl  = (float*)alloc(M * 320  * sizeof(float));
    float*    delta = (float*)alloc(M * 1024 * sizeof(float));
    float*    ys    = (float*)alloc(M * 1024 * sizeof(float));
    float*    houts = (float*)alloc((size_t)B * 128 * sizeof(float));
    _Float16* wt16  = (_Float16*)alloc((size_t)1024 * 2048 * sizeof(_Float16));

    auto wcast = [&](const float* w, int K, int N) {
        long total = (long)K * N;
        k_wcast_t<<<dim3((unsigned)((total + 255) / 256)), 256, 0, stream>>>(
            w, wt16, K, N, total);
    };
    auto gemm = [&](const float* A, int lda, float* C, int ldc,
                    const float* bias, const float* res, int ldr,
                    int Mm, int Nn, int Kk, int act) {
        const int nfull = Nn / 64;
        if (nfull > 0) {
            k_wmma_gemm<4><<<dim3(Mm / 16, nfull), dim3(32), 0, stream>>>(
                A, lda, wt16, C, ldc, bias, res, ldr, 0, Kk, act);
        }
        const int rem = Nn - nfull * 64;  // multiple of 16
        if (rem > 0) {
            const int ncol0 = nfull * 64;
            switch (rem >> 4) {
                case 1:
                    k_wmma_gemm<1><<<dim3(Mm / 16, 1), dim3(32), 0, stream>>>(
                        A, lda, wt16, C, ldc, bias, res, ldr, ncol0, Kk, act);
                    break;
                case 2:
                    k_wmma_gemm<2><<<dim3(Mm / 16, 1), dim3(32), 0, stream>>>(
                        A, lda, wt16, C, ldc, bias, res, ldr, ncol0, Kk, act);
                    break;
                default:
                    k_wmma_gemm<3><<<dim3(Mm / 16, 1), dim3(32), 0, stream>>>(
                        A, lda, wt16, C, ldc, bias, res, ldr, ncol0, Kk, act);
                    break;
            }
        }
    };

    for (int head = 0; head < 2; ++head) {
        const int base = 1 + head * 13;
        const float* win   = (const float*)d_in[base + 0];
        const float* bin   = (const float*)d_in[base + 1];
        const float* wip   = (const float*)d_in[base + 2];
        const float* cw    = (const float*)d_in[base + 3];
        const float* cb    = (const float*)d_in[base + 4];
        const float* wx    = (const float*)d_in[base + 5];
        const float* wdt   = (const float*)d_in[base + 6];
        const float* bdt   = (const float*)d_in[base + 7];
        const float* alog  = (const float*)d_in[base + 8];
        const float* dskip = (const float*)d_in[base + 9];
        const float* wop   = (const float*)d_in[base + 10];
        const float* wol   = (const float*)d_in[base + 11];
        const float* bol   = (const float*)d_in[base + 12];

        const int d  = (head == 0) ? 512 : 1024;
        const int r  = (d + 15) / 16;       // 32 / 64
        const int xw = r + 2 * NST;         // 288 / 320

        // 1) tmp = x @ win + bin
        wcast(win, IN, d);
        gemm(x, IN, tmp, d, bin, nullptr, 0, (int)M, d, IN, 0);

        // 2) xz = tmp @ wip
        wcast(wip, d, 2 * d);
        gemm(tmp, d, xz, 2 * d, nullptr, nullptr, 0, (int)M, 2 * d, d, 0);

        // 3) causal conv(k=2) + silu -> xc
        {
            long total = M * d;
            k_conv_silu<<<dim3((unsigned)((total + 255) / 256)), 256, 0, stream>>>(
                xz, xc, cw, cb, d, L, total);
        }

        // 4) x_dbl = xc @ wx
        wcast(wx, d, xw);
        gemm(xc, d, xdbl, xw, nullptr, nullptr, 0, (int)M, xw, d, 0);

        // 5) delta = softplus(dtp @ wdt + bdt)   (dtp = first r cols of x_dbl)
        wcast(wdt, r, d);
        gemm(xdbl, xw, delta, d, bdt, nullptr, 0, (int)M, d, r, 1);

        // 6) selective scan -> ys (async LDS double-buffered B/C rows)
        k_scan<<<dim3(B * d), dim3(128), 0, stream>>>(
            xc, delta, xdbl, xw, r, alog, dskip, xz, 2 * d, ys, d, L, NST);

        // 7) tmp = tmp + ys @ wop  (residual fused, in place)
        wcast(wop, d, d);
        gemm(ys, d, tmp, d, nullptr, tmp, d, (int)M, d, d, 0);

        // 8) last token only: houts[:, head*64:+64] = tmp[:, L-1, :] @ wol + bol
        wcast(wol, d, OUT);
        gemm(tmp + (long)(L - 1) * d, L * d, houts + head * OUT, 2 * OUT,
             bol, nullptr, 0, B, OUT, d, 0);
    }

    // final: out = houts @ wg + bg  -> [B,1]
    k_final<<<dim3(1), dim3(32), 0, stream>>>(
        houts, (const float*)d_in[27], (const float*)d_in[28], (float*)d_out, B, 2 * OUT);
}
